// GaussianEncoder_26285199851904
// MI455X (gfx1250) — matrix operand order repaired and checked by reading, not compiled, not run on hardware
//
#include <hip/hip_runtime.h>
#include <hip/hip_bf16.h>
#include <math.h>

typedef __attribute__((ext_vector_type(16))) _Float16 v16h;
typedef __attribute__((ext_vector_type(8)))  float    v8f;

#define S 32            // STATE_DIM
#define NODE_DIM 7

// ---------------------------------------------------------------------------
// state = relu(x @ W_in + b_in)   (N,7)@(7,32)
// one thread per (node, channel); W_in column reads coalesced across lanes
// ---------------------------------------------------------------------------
__global__ __launch_bounds__(256) void input_proj_kernel(
    const float* __restrict__ x, const float* __restrict__ W_in,
    const float* __restrict__ b_in, float* __restrict__ state, int N)
{
    long long tid = (long long)blockIdx.x * blockDim.x + threadIdx.x;
    if (tid >= (long long)N * S) return;
    int n = (int)(tid >> 5);
    int c = (int)(tid & 31);
    float acc = b_in[c];
    #pragma unroll
    for (int k = 0; k < NODE_DIM; ++k)
        acc = fmaf(x[(long long)n * NODE_DIM + k], W_in[k * S + c], acc);
    state[tid] = acc > 0.f ? acc : 0.f;
}

// ---------------------------------------------------------------------------
// Fused (N,32)@(32,32) GEMM via v_wmma_f32_16x16x32_f16.
// One wave32 computes a 16x32 output slab: A tile loaded once (f32->f16 pack
// per ISA 16-bit A layout), two B halves, two WMMAs, bias pre-seeded in C,
// ReLU (+ optional residual) fused into the store.
//   mode 0: out = relu(A@W + b)                (message net)
//   mode 1: out = resid + relu(A@W + b)        (update net, in-place safe)
// ---------------------------------------------------------------------------
__global__ __launch_bounds__(256) void gemm32_wmma_kernel(
    const float* __restrict__ A, const float* __restrict__ W,
    const float* __restrict__ bias, const float* __restrict__ resid,
    float* __restrict__ out, int N, int mode)
{
    int lane  = threadIdx.x & 31;
    int wave  = threadIdx.x >> 5;
    int tile  = blockIdx.x * (blockDim.x >> 5) + wave;
    int m0    = tile * 16;
    if (m0 >= N) return;           // wave-uniform: EXEC stays all-ones for WMMA

    int lhalf = lane >> 4;         // lane group 0..15 vs 16..31
    int lmod  = lane & 15;

    // ---- A: 16x32 f16, ISA layout: lane holds row M=lmod; VGPR v pair holds
    //      K = (v<4 ? 0 : 16) + (lhalf ? 8 : 0) + 2*(v&3) .. +1
    const float* arow = A + (long long)(m0 + lmod) * S;
    v16h a;
    #pragma unroll
    for (int v = 0; v < 8; ++v) {
        int kb = ((v < 4) ? 0 : 16) + (lhalf ? 8 : 0) + 2 * (v & 3);
        a[2 * v]     = (_Float16)arow[kb];
        a[2 * v + 1] = (_Float16)arow[kb + 1];
    }

    // ---- B: 32x16 f16 per half, ISA layout: lane holds col n=lmod;
    //      VGPR v pair holds K = (v<4 ? 0 : 16) + (lhalf ? 8 : 0) + 2*(v&3) .. +1.  W is row-major (K,Ncols=32).
    v16h b0, b1;
    #pragma unroll
    for (int v = 0; v < 8; ++v) {
        int k = ((v < 4) ? 0 : 16) + (lhalf ? 8 : 0) + 2 * (v & 3);
        b0[2 * v]     = (_Float16)W[k * S + lmod];
        b0[2 * v + 1] = (_Float16)W[(k + 1) * S + lmod];
        b1[2 * v]     = (_Float16)W[k * S + 16 + lmod];
        b1[2 * v + 1] = (_Float16)W[(k + 1) * S + 16 + lmod];
    }

    // ---- C seeded with bias (C/D layout: VGPR r -> M = r or 8+r, N = lmod)
    float bv0 = bias[lmod];
    float bv1 = bias[16 + lmod];
    v8f c0, c1;
    #pragma unroll
    for (int r = 0; r < 8; ++r) { c0[r] = bv0; c1[r] = bv1; }

    c0 = __builtin_amdgcn_wmma_f32_16x16x32_f16(false, a, false, b0,
                                                (short)0, c0, false, false);
    c1 = __builtin_amdgcn_wmma_f32_16x16x32_f16(false, a, false, b1,
                                                (short)0, c1, false, false);

    #pragma unroll
    for (int r = 0; r < 8; ++r) {
        int m = m0 + (lhalf ? 8 + r : r);
        float v0 = c0[r] > 0.f ? c0[r] : 0.f;
        float v1 = c1[r] > 0.f ? c1[r] : 0.f;
        long long base = (long long)m * S;
        if (mode == 1) {
            v0 += resid[base + lmod];
            v1 += resid[base + 16 + lmod];
        }
        out[base + lmod]      = v0;
        out[base + 16 + lmod] = v1;
    }
}

// ---------------------------------------------------------------------------
// Edge gather + scatter-add: one wave per edge, one lane per channel.
// 128B coalesced gather from message[src], f32 atomics into agg[dst].
// All buffers fit in the 192MB L2, so atomics resolve at L2 rate.
// ---------------------------------------------------------------------------
__global__ __launch_bounds__(256) void edge_scatter_kernel(
    const float* __restrict__ msg, const int* __restrict__ src,
    const int* __restrict__ dst, float* __restrict__ agg, int E)
{
    long long tid = (long long)blockIdx.x * blockDim.x + threadIdx.x;
    long long e   = tid >> 5;
    if (e >= E) return;
    int lane = (int)(tid & 31);
    int s = src[e];
    int d = dst[e];
    float v = msg[(long long)s * S + lane];
    atomicAdd(&agg[(long long)d * S + lane], v);
}

// ---------------------------------------------------------------------------
// graph_state[batch[n]] += state[n]  (segment_sum over sorted batch ids)
// ---------------------------------------------------------------------------
__global__ __launch_bounds__(256) void pool_kernel(
    const float* __restrict__ state, const int* __restrict__ batch,
    float* __restrict__ graph, int N)
{
    long long tid = (long long)blockIdx.x * blockDim.x + threadIdx.x;
    if (tid >= (long long)N * S) return;
    int n = (int)(tid >> 5);
    int c = (int)(tid & 31);
    int g = batch[n];
    atomicAdd(&graph[(long long)g * S + c], state[tid]);
}

// ---------------------------------------------------------------------------
// out = graph @ W_out + b_out; cols 0..1 = mean, cols 2..3 = exp(log_std)
// ---------------------------------------------------------------------------
__global__ __launch_bounds__(256) void head_kernel(
    const float* __restrict__ graph, const float* __restrict__ W_out,
    const float* __restrict__ b_out, float* __restrict__ out, int G, int OC)
{
    int tid = blockIdx.x * blockDim.x + threadIdx.x;
    if (tid >= G * OC) return;
    int g = tid / OC;
    int j = tid % OC;
    float acc = b_out[j];
    #pragma unroll
    for (int k = 0; k < S; ++k)
        acc = fmaf(graph[g * S + k], W_out[k * OC + j], acc);
    out[tid] = (j < OC / 2) ? acc : expf(acc);
}

// ---------------------------------------------------------------------------
extern "C" void kernel_launch(void* const* d_in, const int* in_sizes, int n_in,
                              void* d_out, int out_size, void* d_ws, size_t ws_size,
                              hipStream_t stream)
{
    const float* x        = (const float*)d_in[0];
    const int*   edge_idx = (const int*)  d_in[1];
    const int*   batch    = (const int*)  d_in[2];
    const float* W_in     = (const float*)d_in[3];
    const float* b_in     = (const float*)d_in[4];
    const float* W_msg    = (const float*)d_in[5];
    const float* b_msg    = (const float*)d_in[6];
    const float* W_upd    = (const float*)d_in[7];
    const float* b_upd    = (const float*)d_in[8];
    const float* W_out    = (const float*)d_in[9];
    const float* b_out    = (const float*)d_in[10];

    const int N      = in_sizes[0] / NODE_DIM;      // 100000
    const int E      = in_sizes[1] / 2;             // 2500000
    const int ROUNDS = in_sizes[5] / (S * S);       // 4
    const int OC     = in_sizes[10];                // 4 (= 2M)
    const int G      = out_size / OC;               // 512

    const int* src = edge_idx;
    const int* dst = edge_idx + E;

    // workspace layout
    float* state = (float*)d_ws;
    float* msg   = state + (size_t)N * S;
    float* agg   = msg   + (size_t)N * S;
    float* graph = agg   + (size_t)N * S;
    (void)ws_size; (void)n_in;

    const int B = 256;

    // 1) input projection
    {
        long long t = (long long)N * S;
        input_proj_kernel<<<(int)((t + B - 1) / B), B, 0, stream>>>(
            x, W_in, b_in, state, N);
    }

    // 2) message-passing rounds
    int tiles      = (N + 15) / 16;
    int gemmBlocks = (tiles + (B / 32) - 1) / (B / 32);
    long long eThreads = (long long)E * S;
    int edgeBlocks = (int)((eThreads + B - 1) / B);

    for (int r = 0; r < ROUNDS; ++r) {
        // message = relu(state @ W_msg[r] + b_msg[r])
        gemm32_wmma_kernel<<<gemmBlocks, B, 0, stream>>>(
            state, W_msg + (size_t)r * S * S, b_msg + (size_t)r * S,
            nullptr, msg, N, 0);

        hipMemsetAsync(agg, 0, (size_t)N * S * sizeof(float), stream);

        // agg[dst] += msg[src]
        edge_scatter_kernel<<<edgeBlocks, B, 0, stream>>>(msg, src, dst, agg, E);

        // state = state + relu(agg @ W_upd[r] + b_upd[r])
        gemm32_wmma_kernel<<<gemmBlocks, B, 0, stream>>>(
            agg, W_upd + (size_t)r * S * S, b_upd + (size_t)r * S,
            state, state, N, 1);
    }

    // 3) graph pooling
    hipMemsetAsync(graph, 0, (size_t)G * S * sizeof(float), stream);
    {
        long long t = (long long)N * S;
        pool_kernel<<<(int)((t + B - 1) / B), B, 0, stream>>>(state, batch, graph, N);
    }

    // 4) Gaussian head
    {
        int t = G * OC;
        head_kernel<<<(t + B - 1) / B, B, 0, stream>>>(graph, W_out, b_out,
                                                       (float*)d_out, G, OC);
    }
}
